// ExpertEngine_3126736191876
// MI455X (gfx1250) — matrix operand order repaired and checked by reading, not compiled, not run on hardware
//
#include <hip/hip_runtime.h>
#include <hip/hip_bf16.h>
#include <cstdint>

// ---------------------------------------------------------------------------
// Problem constants (from reference)
// ---------------------------------------------------------------------------
#define NE      16          // experts
#define NEMBD   1024        // hidden
#define EDIM    2048        // expert dim
#define NTOK    8192        // tokens
#define TOPK    1024        // tokens per expert

typedef __attribute__((ext_vector_type(16))) __bf16        v16bf;
typedef __attribute__((ext_vector_type(8)))  float         v8f;
typedef __attribute__((ext_vector_type(8)))  unsigned int  u32x8;
typedef __attribute__((ext_vector_type(4)))  unsigned int  u32x4;
typedef __attribute__((ext_vector_type(8)))  int           i32x8;
typedef __attribute__((ext_vector_type(4)))  int           i32x4;

// ---------------------------------------------------------------------------
// Kernel 1: router logits + softmax + fanout zeroing. One wave32 per token.
// ---------------------------------------------------------------------------
__global__ __launch_bounds__(256) void router_kernel(
    const float* __restrict__ x, const float* __restrict__ rw,
    float* __restrict__ logits, float* __restrict__ smax,
    float* __restrict__ fanout)
{
    const int wave = threadIdx.x >> 5;
    const int lane = threadIdx.x & 31;
    const int tok  = blockIdx.x * 8 + wave;

    float acc[NE];
#pragma unroll
    for (int e = 0; e < NE; ++e) acc[e] = 0.f;

    const float* xr = x + (size_t)tok * NEMBD;
    for (int f = lane; f < NEMBD; f += 32) {
        const float xv = xr[f];
#pragma unroll
        for (int e = 0; e < NE; ++e) acc[e] += xv * rw[e * NEMBD + f];
    }
#pragma unroll
    for (int e = 0; e < NE; ++e) {
#pragma unroll
        for (int off = 16; off >= 1; off >>= 1)
            acc[e] += __shfl_xor(acc[e], off, 32);
    }

    if (lane == 0) {
        float m = acc[0];
#pragma unroll
        for (int e = 1; e < NE; ++e) m = fmaxf(m, acc[e]);
        float s = 0.f, w[NE];
#pragma unroll
        for (int e = 0; e < NE; ++e) { w[e] = __expf(acc[e] - m); s += w[e]; }
        const float inv = 1.f / s;
#pragma unroll
        for (int e = 0; e < NE; ++e) {
            logits[tok * NE + e] = acc[e];
            smax[tok * NE + e]   = w[e] * inv;
        }
        fanout[tok] = 0.f;
    }
}

// ---------------------------------------------------------------------------
// Kernel 2: exact per-expert top-K via bitonic sort of 8192 (val,idx) pairs
// in 64 KB LDS. Descending value, ascending index tie-break (jax top_k order).
// ---------------------------------------------------------------------------
__global__ __launch_bounds__(1024) void topk_kernel(
    const float* __restrict__ logits, const float* __restrict__ smax,
    int* __restrict__ ws_idx, float* __restrict__ out_idx,
    float* __restrict__ out_w, float* __restrict__ out_fanout,
    float* __restrict__ out_cut)
{
    __shared__ float sv[NTOK];
    __shared__ int   si[NTOK];

    const int e = blockIdx.x;
    const int t = threadIdx.x;

    for (int i = t; i < NTOK; i += 1024) {
        sv[i] = logits[i * NE + e];
        si[i] = i;
    }
    __syncthreads();

    for (int k = 2; k <= NTOK; k <<= 1) {
        for (int j = k >> 1; j > 0; j >>= 1) {
            for (int p = t; p < (NTOK / 2); p += 1024) {
                const int i   = ((p / j) * (j << 1)) + (p % j);
                const int ixj = i + j;
                const float va = sv[i], vb = sv[ixj];
                const int   ia = si[i], ib = si[ixj];
                const bool aGreater = (va > vb) || (va == vb && ia < ib);
                const bool descDir  = ((i & k) == 0);
                if (descDir ? !aGreater : aGreater) {
                    sv[i] = vb; sv[ixj] = va;
                    si[i] = ib; si[ixj] = ia;
                }
            }
            __syncthreads();
        }
    }

    const int idx = si[t];
    ws_idx[e * TOPK + t]  = idx;
    out_idx[e * TOPK + t] = (float)idx;
    out_w[e * TOPK + t]   = smax[idx * NE + e];
    atomicAdd(&out_fanout[idx], 1.0f);
    if (t == TOPK - 1) out_cut[e] = sv[t];
}

// ---------------------------------------------------------------------------
// Kernel 3: gather selected token rows, f32 -> bf16 (WMMA A operand feed).
// ---------------------------------------------------------------------------
__global__ __launch_bounds__(256) void gather_kernel(
    const float* __restrict__ x, const int* __restrict__ ws_idx,
    __bf16* __restrict__ xb)
{
    const int row = blockIdx.x;
    const int idx = ws_idx[row];
    const float4* src = reinterpret_cast<const float4*>(x + (size_t)idx * NEMBD);
    __bf16* dst = xb + (size_t)row * NEMBD;
    const float4 v = src[threadIdx.x];
    const int f = threadIdx.x * 4;
    dst[f + 0] = (__bf16)v.x;
    dst[f + 1] = (__bf16)v.y;
    dst[f + 2] = (__bf16)v.z;
    dst[f + 3] = (__bf16)v.w;
}

// ---------------------------------------------------------------------------
// TDM issue helper: 2D tile (tile_dim0=32 bf16 along K, tile_dim1=128 rows),
// padded LDS rows (16 dwords data + 2 dwords pad = 72B stride).
// D# layout per CDNA5 ISA.
// ---------------------------------------------------------------------------
__device__ __forceinline__ void tdm_load_tile_a(
    unsigned lds_byte, unsigned long long gaddr, int KC, int M)
{
    u32x4 g0;
    g0[0] = 1u;                                    // count=1 (valid user D#)
    g0[1] = lds_byte;                              // lds_addr
    g0[2] = (unsigned)(gaddr & 0xffffffffull);     // global_addr[31:0]
    g0[3] = (unsigned)((gaddr >> 32) & 0x1ffffffull) | (2u << 30); // addr[56:32] | type=2

    i32x8 g1;
    // data_size=2B (1<<16) | pad_enable (1<<20) | pad_interval=16dw (3<<22)
    // | pad_amount=2dw (1<<25)
    g1[0] = 0x02D10000;
    g1[1] = (int)((KC & 0xffff) << 16);                    // tensor_dim0[15:0]
    g1[2] = (int)((KC >> 16) & 0xffff) | ((M & 0xffff) << 16); // td0 hi | td1 lo
    g1[3] = (int)((M >> 16) & 0xffff) | (32 << 16);        // td1 hi | tile_dim0=32
    g1[4] = 128;                                           // tile_dim1=128, tile_dim2=0
    g1[5] = KC;                                            // tensor_dim0_stride lo
    g1[6] = 0;                                             // stride0 hi | stride1 lo
    g1[7] = 0;

    i32x4 z4 = {0, 0, 0, 0};
#if __has_include(<hip/amd_detail/amd_gfx1250_TDM.h>)
    i32x8 z8 = {0, 0, 0, 0, 0, 0, 0, 0};
    __builtin_amdgcn_tensor_load_to_lds(g0, g1, z4, z4, z8, 0);
#else
    __builtin_amdgcn_tensor_load_to_lds(g0, g1, z4, z4, 0);
#endif
}

// ---------------------------------------------------------------------------
// Kernels 4/5: tiled bf16 WMMA GEMM, out[e] = A[e] (MxKC bf16) x W[e]^T.
// Software-pipelined, ping-pong LDS:
//   - TDM for A tile t+1 is issued before computing tile t (wave 0 only),
//     drained with s_wait_tensorcnt 1 (in-order per-wave completion).
//   - W tile t+1 is fetched to registers during compute of tile t; its
//     f32->bf16 K-pair packing happens at the top of the next iteration.
// Block tile 128x128x32; 8 waves, each owns a 32x64 macro-tile (2x4 WMMAs).
// ---------------------------------------------------------------------------
template <int M, int N, int KC, bool ACT>
__global__ __launch_bounds__(256) void moe_gemm(
    const __bf16* __restrict__ A, const float* __restrict__ W,
    __bf16* __restrict__ outb, float* __restrict__ outf)
{
    __shared__ __bf16       As[2][128][36];   // row stride 72B (TDM pad-matched)
    __shared__ unsigned int Bs32[2][16][132]; // [k-pair][n] packed bf16x2 dwords

    const int e  = blockIdx.z;
    const int m0 = blockIdx.x * 128;
    const int n0 = blockIdx.y * 128;

    A += (size_t)e * M * KC;
    W += (size_t)e * N * KC;

    const int tid  = threadIdx.x;
    const int lane = tid & 31;
    const int lrow = lane & 15;
    const int lhi  = lane >> 4;
    const int wv   = tid >> 5;
    const int wm   = wv & 3;               // 4 waves along M
    const int wn   = wv >> 2;              // 2 waves along N

    const int bn = tid & 127;              // W loader: one n-column
    const int bk = (tid >> 7) * 16;        // 16 consecutive k per thread

    const unsigned lds_a0    = (unsigned)(size_t)(&As[0][0][0]);
    const unsigned lds_a_sz  = 128 * 36 * 2;
    const unsigned long long abase =
        (unsigned long long)(size_t)(const void*)(A + (size_t)m0 * KC);
    const float* wrow = W + (size_t)(n0 + bn) * KC + bk;

    constexpr int T = KC / 32;             // k-tiles
    v8f acc[2][4] = {};
    float4 breg[4];

    // ---- prologue: stage tile 0 ----
    if (wv == 0) tdm_load_tile_a(lds_a0, abase, KC, M);
#pragma unroll
    for (int i4 = 0; i4 < 4; ++i4)
        breg[i4] = reinterpret_cast<const float4*>(wrow)[i4];

    for (int t = 0; t < T; ++t) {
        const int cur = t & 1;
        const int k0n = (t + 1) * 32;

        // ---- pack W tile t (in registers) into LDS K-pair dwords ----
#pragma unroll
        for (int i4 = 0; i4 < 4; ++i4) {
            const float4 v = breg[i4];
            union { unsigned u; __bf16 b[2]; } p0, p1;
            p0.b[0] = (__bf16)v.x; p0.b[1] = (__bf16)v.y;
            p1.b[0] = (__bf16)v.z; p1.b[1] = (__bf16)v.w;
            Bs32[cur][(bk >> 1) + i4 * 2 + 0][bn] = p0.u;
            Bs32[cur][(bk >> 1) + i4 * 2 + 1][bn] = p1.u;
        }

        // ---- issue next tile's TDM + W global loads (overlap compute) ----
        if (t + 1 < T) {
            if (wv == 0)
                tdm_load_tile_a(lds_a0 + ((t + 1) & 1) * lds_a_sz,
                                abase + (unsigned long long)k0n * 2, KC, M);
#pragma unroll
            for (int i4 = 0; i4 < 4; ++i4)
                breg[i4] = reinterpret_cast<const float4*>(wrow + k0n)[i4];
            if (t + 2 < T)
                __builtin_prefetch(wrow + (t + 2) * 32, 0, 0);
        }

        // ---- drain TDM for tile t (allow tile t+1 in flight), sync ----
        if (wv == 0) {
            if (t + 1 < T) __builtin_amdgcn_s_wait_tensorcnt(1);
            else           __builtin_amdgcn_s_wait_tensorcnt(0);
        }
        __syncthreads();

        // ---- fragments (contiguous k = 16*lhi + s in A and B), dword LDS reads + bit_cast ----
        v16bf afrag[2];
#pragma unroll
        for (int mi = 0; mi < 2; ++mi) {
            const int mr = wm * 32 + mi * 16 + lrow;
            u32x8 ar;
#pragma unroll
            for (int v = 0; v < 8; ++v) {
                const int kk = 2 * v + lhi * 16; // even
                ar[v] = *reinterpret_cast<const unsigned int*>(&As[cur][mr][kk]);
            }
            afrag[mi] = __builtin_bit_cast(v16bf, ar);
        }
        v16bf bfrag[4];
#pragma unroll
        for (int ni = 0; ni < 4; ++ni) {
            const int nc = wn * 64 + ni * 16 + lrow;
            u32x8 br;
#pragma unroll
            for (int v = 0; v < 8; ++v)
                br[v] = Bs32[cur][v + lhi * 8][nc];
            bfrag[ni] = __builtin_bit_cast(v16bf, br);
        }

#pragma unroll
        for (int mi = 0; mi < 2; ++mi)
#pragma unroll
            for (int ni = 0; ni < 4; ++ni)
                acc[mi][ni] = __builtin_amdgcn_wmma_f32_16x16x32_bf16(
                    false, afrag[mi], false, bfrag[ni],
                    (short)0, acc[mi][ni], false, false);

        // all waves done reading buffers before wave0 can overwrite them
        __syncthreads();
    }

    // ---- epilogue ----
#pragma unroll
    for (int mi = 0; mi < 2; ++mi) {
#pragma unroll
        for (int ni = 0; ni < 4; ++ni) {
            const int mbase = m0 + wm * 32 + mi * 16;
            const int nc    = n0 + wn * 64 + ni * 16 + lrow;
#pragma unroll
            for (int v = 0; v < 8; ++v) {
                const int mr = mbase + v + lhi * 8;   // C/D layout: M = v + half*8
                const float val = acc[mi][ni][v];
                if (ACT) {
                    const float r = fmaxf(val, 0.f);
                    outb[((size_t)e * M + mr) * N + nc] = (__bf16)(r * r);
                } else {
                    outf[((size_t)e * M + mr) * N + nc] = val;
                }
            }
        }
    }
}

// ---------------------------------------------------------------------------
// Launcher
// ---------------------------------------------------------------------------
extern "C" void kernel_launch(void* const* d_in, const int* in_sizes, int n_in,
                              void* d_out, int out_size, void* d_ws, size_t ws_size,
                              hipStream_t stream)
{
    (void)in_sizes; (void)n_in; (void)out_size; (void)ws_size;

    const float* x  = (const float*)d_in[0];
    const float* rw = (const float*)d_in[1];
    const float* w1 = (const float*)d_in[2];
    const float* w2 = (const float*)d_in[3];

    // d_out layout: h_flat | indices | weights | fanout | cutoffs (all f32)
    float* out_h      = (float*)d_out;
    float* out_idx    = out_h + (size_t)NE * TOPK * NEMBD;
    float* out_w      = out_idx + NE * TOPK;
    float* out_fanout = out_w + NE * TOPK;
    float* out_cut    = out_fanout + NTOK;

    // workspace layout
    char* ws = (char*)d_ws;
    float*  logits = (float*)ws;                         ws += (size_t)NTOK * NE * 4;
    float*  smax   = (float*)ws;                         ws += (size_t)NTOK * NE * 4;
    int*    idxbuf = (int*)ws;                           ws += (size_t)NE * TOPK * 4;
    __bf16* xb     = (__bf16*)ws;                        ws += (size_t)NE * TOPK * NEMBD * 2;
    __bf16* hbuf   = (__bf16*)ws;                        // NE*TOPK*EDIM bf16

    router_kernel<<<NTOK / 8, 256, 0, stream>>>(x, rw, logits, smax, out_fanout);
    topk_kernel<<<NE, 1024, 0, stream>>>(logits, smax, idxbuf, out_idx, out_w,
                                         out_fanout, out_cut);
    gather_kernel<<<NE * TOPK, 256, 0, stream>>>(x, idxbuf, xb);

    moe_gemm<TOPK, EDIM, NEMBD, true>
        <<<dim3(TOPK / 128, EDIM / 128, NE), 256, 0, stream>>>(xb, w1, hbuf, nullptr);
    moe_gemm<TOPK, NEMBD, EDIM, false>
        <<<dim3(TOPK / 128, NEMBD / 128, NE), 256, 0, stream>>>(hbuf, w2, nullptr, out_h);
}
